// NodeEdgeAggregatorV3_58944131170469
// MI455X (gfx1250) — compile-verified, matrix operand order checked by reading, not run on hardware
//
#include <hip/hip_runtime.h>
#include <hip/hip_bf16.h>
#include <math.h>

// ---------------------------------------------------------------------------
// Types for CDNA5 WMMA (wave32). bf16 A/B, f32 accumulate.
// ---------------------------------------------------------------------------
typedef __attribute__((ext_vector_type(16))) __bf16 v16bf;
typedef __attribute__((ext_vector_type(4)))  __bf16 v4bf;
typedef __attribute__((ext_vector_type(8)))  float  v8f;

#define HID 256
#define TM 128  // block tile rows
#define TN 64   // block tile cols
#define TK 32   // k-step (matches 16x16x32 bf16 WMMA)

// Fragment swizzle (ISA 16-bit A/B layout):
//   lane = (idx&15) + 16*((k>>3)&1)      [idx = row for A, col for B]
//   elem = (k&7) + 8*(k>>4)
// -> each lane's 16 fragment elements are contiguous in LDS (2x ds_load_b128)
// -> 4 k-contiguous (4-aligned) values are contiguous elems (1x ds_store_b64)

// ---------------------------------------------------------------------------
// Tiled GEMM: C[M,N] = A[M,K] @ B[K,N] (+ bias[N]) (+ C if addC)
// 256 threads = 8 waves. Block tile 128x64. Wave w owns a 32x32 output:
// row tiles 2*(w&3)+{0,1}, col tiles 2*(w>>2)+{0,1} -> 4 WMMAs/K-step with
// full A/B fragment reuse (2x2 register blocking).
// Double-buffered LDS; register-staged prefetch of next K-step.
// K multiple of 32. M arbitrary (clamped loads). NFULL: N multiple of 64.
// ---------------------------------------------------------------------------
template <bool NFULL>
__global__ __launch_bounds__(256) void gemm_wmma_kernel(
    const float* __restrict__ A, const float* __restrict__ B,
    float* __restrict__ C, const float* __restrict__ bias,
    int M, int N, int K, int addC)
{
  __shared__ __align__(32) __bf16 Afrag[2][8][32][16];  // [buf][rowtile][lane][elem]
  __shared__ __align__(32) __bf16 Bfrag[2][4][32][16];  // [buf][coltile][lane][elem]

  const int tid  = threadIdx.x;
  const int bm   = blockIdx.y * TM;
  const int bn   = blockIdx.x * TN;
  const int wave = tid >> 5;
  const int lane = tid & 31;
  const int half = lane >> 4;
  const int l15  = lane & 15;
  const int wrow = (wave & 3) * 2;    // first row tile (0,2,4,6)
  const int wcol = (wave >> 2) * 2;   // first col tile (0 or 2)

  // --- register-staged tile fetch (straight-line, clamped addresses) ---
  float4 ra[4];
  float  rb[2][4];

  auto loadTiles = [&](int k0) {
    #pragma unroll
    for (int it = 0; it < 4; ++it) {
      int i  = tid + it * 256;
      int m  = i >> 3;                 // 0..127
      int kq = (i & 7) * 4;            // 0..28
      int gm = bm + m;
      int gmc = gm < M ? gm : M - 1;   // clamp: always-in-bounds load
      float4 v = *(const float4*)(A + (size_t)gmc * K + (k0 + kq));
      if (gm >= M) v = make_float4(0.f, 0.f, 0.f, 0.f);
      ra[it] = v;
    }
    #pragma unroll
    for (int it = 0; it < 2; ++it) {
      int n  = tid & 63;
      int kq = (tid >> 6) * 4 + it * 16;
      int gn = bn + n;
      if (NFULL) {
        const float* bp = B + (size_t)(k0 + kq) * N + gn;
        #pragma unroll
        for (int j = 0; j < 4; ++j) rb[it][j] = bp[(size_t)j * N];
      } else {
        int gnc = gn < N ? gn : N - 1;
        const float* bp = B + (size_t)(k0 + kq) * N + gnc;
        #pragma unroll
        for (int j = 0; j < 4; ++j) rb[it][j] = (gn < N) ? bp[(size_t)j * N] : 0.f;
      }
    }
  };

  auto storeTiles = [&](int buf) {
    #pragma unroll
    for (int it = 0; it < 4; ++it) {
      int i  = tid + it * 256;
      int m  = i >> 3;
      int kq = (i & 7) * 4;
      int ln = (m & 15) + 16 * ((kq >> 3) & 1);
      int e  = (kq & 7) + 8 * (kq >> 4);
      v4bf w;
      w[0] = (__bf16)ra[it].x; w[1] = (__bf16)ra[it].y;
      w[2] = (__bf16)ra[it].z; w[3] = (__bf16)ra[it].w;
      *(v4bf*)&Afrag[buf][m >> 4][ln][e] = w;       // ds_store_b64
    }
    #pragma unroll
    for (int it = 0; it < 2; ++it) {
      int n  = tid & 63;
      int kq = (tid >> 6) * 4 + it * 16;
      int ln = (n & 15) + 16 * ((kq >> 3) & 1);
      int e  = (kq & 7) + 8 * (kq >> 4);
      v4bf w;
      w[0] = (__bf16)rb[it][0]; w[1] = (__bf16)rb[it][1];
      w[2] = (__bf16)rb[it][2]; w[3] = (__bf16)rb[it][3];
      *(v4bf*)&Bfrag[buf][n >> 4][ln][e] = w;       // ds_store_b64
    }
  };

  v8f acc00 = {}, acc01 = {}, acc10 = {}, acc11 = {};

  const int steps = K / TK;
  loadTiles(0);
  storeTiles(0);
  __syncthreads();

  for (int kt = 0; kt < steps; ++kt) {
    int cur = kt & 1;
    if (kt + 1 < steps) loadTiles((kt + 1) * TK);   // prefetch overlaps WMMA

    v16bf a0 = *(const v16bf*)&Afrag[cur][wrow][lane][0];
    v16bf a1 = *(const v16bf*)&Afrag[cur][wrow + 1][lane][0];
    v16bf b0 = *(const v16bf*)&Bfrag[cur][wcol][lane][0];
    v16bf b1 = *(const v16bf*)&Bfrag[cur][wcol + 1][lane][0];
    acc00 = __builtin_amdgcn_wmma_f32_16x16x32_bf16(false, a0, false, b0,
                                                    (short)0, acc00, false, false);
    acc01 = __builtin_amdgcn_wmma_f32_16x16x32_bf16(false, a0, false, b1,
                                                    (short)0, acc01, false, false);
    acc10 = __builtin_amdgcn_wmma_f32_16x16x32_bf16(false, a1, false, b0,
                                                    (short)0, acc10, false, false);
    acc11 = __builtin_amdgcn_wmma_f32_16x16x32_bf16(false, a1, false, b1,
                                                    (short)0, acc11, false, false);
    if (kt + 1 < steps) {
      storeTiles(cur ^ 1);
      __syncthreads();
    }
  }

  // C/D layout: VGPR r -> M = r + 8*half, N = lane&15
  v8f* accs[2][2] = {{&acc00, &acc01}, {&acc10, &acc11}};
  #pragma unroll
  for (int tr = 0; tr < 2; ++tr) {
    #pragma unroll
    for (int tc = 0; tc < 2; ++tc) {
      v8f& acc = *accs[tr][tc];
      int gn = bn + (wcol + tc) * 16 + l15;
      if (!NFULL && gn >= N) continue;
      float bv = bias ? bias[gn] : 0.0f;
      #pragma unroll
      for (int r = 0; r < 8; ++r) {
        int gm = bm + (wrow + tr) * 16 + r + 8 * half;
        if (gm >= M) continue;
        float v = acc[r] + bv;
        size_t o = (size_t)gm * N + gn;
        if (addC) v += C[o];
        C[o] = v;
      }
    }
  }
}

// ---------------------------------------------------------------------------
// Elementwise / utility kernels
// ---------------------------------------------------------------------------
__global__ void fill_kernel(float* p, float v, size_t n) {
  for (size_t i = (size_t)blockIdx.x * blockDim.x + threadIdx.x; i < n;
       i += (size_t)gridDim.x * blockDim.x) p[i] = v;
}

__global__ void bcast_rows_kernel(float* out, const float* __restrict__ bias, size_t rows) {
  size_t total = rows * HID;
  for (size_t i = (size_t)blockIdx.x * blockDim.x + threadIdx.x; i < total;
       i += (size_t)gridDim.x * blockDim.x) out[i] = bias[i & (HID - 1)];
}

__global__ void relu_kernel(float4* p, size_t n4) {
  for (size_t i = (size_t)blockIdx.x * blockDim.x + threadIdx.x; i < n4;
       i += (size_t)gridDim.x * blockDim.x) {
    float4 v = p[i];
    v.x = fmaxf(v.x, 0.f); v.y = fmaxf(v.y, 0.f);
    v.z = fmaxf(v.z, 0.f); v.w = fmaxf(v.w, 0.f);
    p[i] = v;
  }
}

// etn = leaky_relu(acc / max(deg,1), 0.2)
__global__ void etn_lrelu_kernel(const float4* __restrict__ acc, const float* __restrict__ deg,
                                 float4* out, size_t rows) {
  size_t total = rows * (HID / 4);
  for (size_t i = (size_t)blockIdx.x * blockDim.x + threadIdx.x; i < total;
       i += (size_t)gridDim.x * blockDim.x) {
    float inv = 1.0f / fmaxf(deg[i >> 6], 1.0f);
    float4 v = acc[i];
    v.x *= inv; v.y *= inv; v.z *= inv; v.w *= inv;
    v.x = v.x > 0.f ? v.x : 0.2f * v.x;
    v.y = v.y > 0.f ? v.y : 0.2f * v.y;
    v.z = v.z > 0.f ? v.z : 0.2f * v.z;
    v.w = v.w > 0.f ? v.w : 0.2f * v.w;
    out[i] = v;
  }
}

__global__ void mean_div_kernel(float4* m, const float* __restrict__ deg, size_t rows) {
  size_t total = rows * (HID / 4);
  for (size_t i = (size_t)blockIdx.x * blockDim.x + threadIdx.x; i < total;
       i += (size_t)gridDim.x * blockDim.x) {
    float inv = 1.0f / fmaxf(deg[i >> 6], 1.0f);
    float4 v = m[i];
    v.x *= inv; v.y *= inv; v.z *= inv; v.w *= inv;
    m[i] = v;
  }
}

// ---------------------------------------------------------------------------
// Wave-per-row dual dot: o1[r]=dot(H[r],a1), o2[r]=dot(H[r],a2)
// ---------------------------------------------------------------------------
__global__ void dual_dot_kernel(const float* __restrict__ Hm, const float* __restrict__ a1,
                                const float* __restrict__ a2, float* o1, float* o2, int rows) {
  int wave = (int)((blockIdx.x * blockDim.x + threadIdx.x) >> 5);
  int lane = threadIdx.x & 31;
  int nw   = (int)((gridDim.x * blockDim.x) >> 5);
  for (int r = wave; r < rows; r += nw) {
    const float4* row = (const float4*)(Hm + (size_t)r * HID);
    float s1 = 0.f, s2 = 0.f;
    #pragma unroll
    for (int j = 0; j < 2; ++j) {
      int c = lane + 32 * j;
      float4 xv = row[c];
      float4 u = ((const float4*)a1)[c];
      float4 w = ((const float4*)a2)[c];
      s1 += xv.x * u.x + xv.y * u.y + xv.z * u.z + xv.w * u.w;
      s2 += xv.x * w.x + xv.y * w.y + xv.z * w.z + xv.w * w.w;
    }
    for (int o = 16; o; o >>= 1) { s1 += __shfl_xor(s1, o, 32); s2 += __shfl_xor(s2, o, 32); }
    if (lane == 0) { o1[r] = s1; o2[r] = s2; }
  }
}

// Wave-per-edge attention logit: sc[e] = dot(q[dst[e]], k[src[e]]) / sqrt(HID)
__global__ void attn_score_kernel(const float* __restrict__ q, const float* __restrict__ k,
                                  const int* __restrict__ src, const int* __restrict__ dst,
                                  float* sc, int nE) {
  int wave = (int)((blockIdx.x * blockDim.x + threadIdx.x) >> 5);
  int lane = threadIdx.x & 31;
  int nw   = (int)((gridDim.x * blockDim.x) >> 5);
  for (int e = wave; e < nE; e += nw) {
    const float4* qr = (const float4*)(q + (size_t)dst[e] * HID);
    const float4* kr = (const float4*)(k + (size_t)src[e] * HID);
    float s = 0.f;
    #pragma unroll
    for (int j = 0; j < 2; ++j) {
      int c = lane + 32 * j;
      float4 a = qr[c], b = kr[c];
      s += a.x * b.x + a.y * b.y + a.z * b.z + a.w * b.w;
    }
    for (int o = 16; o; o >>= 1) s += __shfl_xor(s, o, 32);
    if (lane == 0) sc[e] = s * 0.0625f;   // 1/sqrt(256)
  }
}

// ---------------------------------------------------------------------------
// GAT scores over line graph + self loops
// ---------------------------------------------------------------------------
__global__ void gat_score_kernel(const float* __restrict__ hs, const float* __restrict__ hd,
                                 const int* __restrict__ lsrc, const int* __restrict__ ldst,
                                 float* sc, int elg, int etot) {
  for (int i = (int)(blockIdx.x * blockDim.x + threadIdx.x); i < etot;
       i += (int)(gridDim.x * blockDim.x)) {
    int s, d;
    if (i < elg) { s = lsrc[i]; d = ldst[i]; } else { s = d = i - elg; }
    float e = hs[s] + hd[d];
    sc[i] = e > 0.0f ? e : 0.2f * e;
  }
}

__device__ inline void atomicMaxF(float* addr, float val) {
  int* ia = (int*)addr;
  int old = __float_as_int(*addr);
  while (__int_as_float(old) < val) {
    int assumed = old;
    old = atomicCAS(ia, assumed, __float_as_int(val));
    if (old == assumed) break;
  }
}

// Segment softmax (3 passes). dst = (i<elg)? ddst[i] : i-elg (self loop).
__global__ void seg_max_kernel(const float* __restrict__ sc, const int* __restrict__ ddst,
                               float* m, int elg, int etot) {
  for (int i = (int)(blockIdx.x * blockDim.x + threadIdx.x); i < etot;
       i += (int)(gridDim.x * blockDim.x)) {
    int d = (i < elg) ? ddst[i] : i - elg;
    atomicMaxF(m + d, sc[i]);
  }
}
__global__ void seg_exp_kernel(float* sc, const int* __restrict__ ddst,
                               const float* __restrict__ m, float* s, int elg, int etot) {
  for (int i = (int)(blockIdx.x * blockDim.x + threadIdx.x); i < etot;
       i += (int)(gridDim.x * blockDim.x)) {
    int d = (i < elg) ? ddst[i] : i - elg;
    float ex = __expf(sc[i] - m[d]);
    sc[i] = ex;
    atomicAdd(s + d, ex);
  }
}
__global__ void seg_norm_kernel(float* sc, const int* __restrict__ ddst,
                                const float* __restrict__ s, int elg, int etot) {
  for (int i = (int)(blockIdx.x * blockDim.x + threadIdx.x); i < etot;
       i += (int)(gridDim.x * blockDim.x)) {
    int d = (i < elg) ? ddst[i] : i - elg;
    sc[i] = sc[i] / (s[d] + 1e-16f);
  }
}

// out[dst[e],:] += alpha[e] * h[src[e],:]   (4 cols per thread)
__global__ void weighted_scatter_kernel(const float* __restrict__ alpha, const float* __restrict__ h,
                                        const int* __restrict__ lsrc, const int* __restrict__ ldst,
                                        float* out, int elg, int etot) {
  size_t total = (size_t)etot * (HID / 4);
  for (size_t i = (size_t)blockIdx.x * blockDim.x + threadIdx.x; i < total;
       i += (size_t)gridDim.x * blockDim.x) {
    int e = (int)(i >> 6), c = (int)(i & 63) * 4;
    int s, d;
    if (e < elg) { s = lsrc[e]; d = ldst[e]; } else { s = d = e - elg; }
    float a = alpha[e];
    float4 v = *(const float4*)(h + (size_t)s * HID + c);
    float* o = out + (size_t)d * HID + c;
    atomicAdd(o + 0, a * v.x);
    atomicAdd(o + 1, a * v.y);
    atomicAdd(o + 2, a * v.z);
    atomicAdd(o + 3, a * v.w);
  }
}

// Incidence scatter: acc[H0[e],:] += he[e,:]; acc[H1[e],:] += he[e,:]
__global__ void inc_scatter_kernel(const float* __restrict__ he, const int* __restrict__ H0,
                                   const int* __restrict__ H1, float* acc, int nE) {
  size_t total = (size_t)nE * (HID / 4);
  for (size_t i = (size_t)blockIdx.x * blockDim.x + threadIdx.x; i < total;
       i += (size_t)gridDim.x * blockDim.x) {
    int e = (int)(i >> 6), c = (int)(i & 63) * 4;
    float4 v = *(const float4*)(he + (size_t)e * HID + c);
    float* o0 = acc + (size_t)H0[e] * HID + c;
    float* o1 = acc + (size_t)H1[e] * HID + c;
    atomicAdd(o0 + 0, v.x); atomicAdd(o0 + 1, v.y);
    atomicAdd(o0 + 2, v.z); atomicAdd(o0 + 3, v.w);
    atomicAdd(o1 + 0, v.x); atomicAdd(o1 + 1, v.y);
    atomicAdd(o1 + 2, v.z); atomicAdd(o1 + 3, v.w);
  }
}
__global__ void inc_deg_kernel(const int* __restrict__ H0, const int* __restrict__ H1,
                               float* deg, int nE) {
  for (int e = (int)(blockIdx.x * blockDim.x + threadIdx.x); e < nE;
       e += (int)(gridDim.x * blockDim.x)) {
    atomicAdd(deg + H0[e], 1.0f);
    atomicAdd(deg + H1[e], 1.0f);
  }
}
__global__ void deg_count_kernel(const int* __restrict__ dst, float* deg, int nE) {
  for (int e = (int)(blockIdx.x * blockDim.x + threadIdx.x); e < nE;
       e += (int)(gridDim.x * blockDim.x)) atomicAdd(deg + dst[e], 1.0f);
}

// mean[dst[e],:] += X[src[e],:]
__global__ void sage_scatter_kernel(const float* __restrict__ X, const int* __restrict__ src,
                                    const int* __restrict__ dst, float* mean, int nE) {
  size_t total = (size_t)nE * (HID / 4);
  for (size_t i = (size_t)blockIdx.x * blockDim.x + threadIdx.x; i < total;
       i += (size_t)gridDim.x * blockDim.x) {
    int e = (int)(i >> 6), c = (int)(i & 63) * 4;
    float4 v = *(const float4*)(X + (size_t)src[e] * HID + c);
    float* o = mean + (size_t)dst[e] * HID + c;
    atomicAdd(o + 0, v.x); atomicAdd(o + 1, v.y);
    atomicAdd(o + 2, v.z); atomicAdd(o + 3, v.w);
  }
}

// log_softmax over rows of 40 cols; wave per row
__global__ void logsoftmax40_kernel(const float* __restrict__ in, float* out, int rows) {
  int wave = (int)((blockIdx.x * blockDim.x + threadIdx.x) >> 5);
  int lane = threadIdx.x & 31;
  int nw   = (int)((gridDim.x * blockDim.x) >> 5);
  for (int r = wave; r < rows; r += nw) {
    const float* row = in + (size_t)r * 40;
    float x1 = row[lane];
    float x2 = (lane < 8) ? row[lane + 32] : -3.0e38f;
    float m = fmaxf(x1, x2);
    for (int o = 16; o; o >>= 1) m = fmaxf(m, __shfl_xor(m, o, 32));
    float s = __expf(x1 - m) + ((lane < 8) ? __expf(x2 - m) : 0.0f);
    for (int o = 16; o; o >>= 1) s += __shfl_xor(s, o, 32);
    float l = m + __logf(s);
    float* orow = out + (size_t)r * 40;
    orow[lane] = x1 - l;
    if (lane < 8) orow[lane + 32] = x2 - l;
  }
}

// ---------------------------------------------------------------------------
// Host-side orchestration
// ---------------------------------------------------------------------------
extern "C" void kernel_launch(void* const* d_in, const int* in_sizes, int n_in,
                              void* d_out, int out_size, void* d_ws, size_t ws_size,
                              hipStream_t stream) {
  const float* x     = (const float*)d_in[0];
  const float* et    = (const float*)d_in[1];
  const int*   Hm    = (const int*)d_in[2];
  const int*   rei   = (const int*)d_in[3];
  const int*   lg    = (const int*)d_in[4];
  const float* W_gat = (const float*)d_in[5];
  const float* a_src = (const float*)d_in[6];
  const float* a_dst = (const float*)d_in[7];
  const float* b_gat = (const float*)d_in[8];
  const float* W_etn = (const float*)d_in[9];
  const float* W_eg  = (const float*)d_in[10];
  const float* Wr_e1 = (const float*)d_in[11];
  const float* Wn_e1 = (const float*)d_in[12];
  const float* b_e1  = (const float*)d_in[13];
  const float* Wr_e2 = (const float*)d_in[14];
  const float* Wn_e2 = (const float*)d_in[15];
  const float* b_e2  = (const float*)d_in[16];
  const float* Wr_n1 = (const float*)d_in[17];
  const float* Wn_n1 = (const float*)d_in[18];
  const float* b_n1  = (const float*)d_in[19];
  const float* Wr_n2 = (const float*)d_in[20];
  const float* Wn_n2 = (const float*)d_in[21];
  const float* b_n2  = (const float*)d_in[22];
  const float* Wr_n3 = (const float*)d_in[23];
  const float* Wn_n3 = (const float*)d_in[24];
  const float* b_n3  = (const float*)d_in[25];
  const float* Wq    = (const float*)d_in[26];
  const float* Wk    = (const float*)d_in[27];
  const float* Wv    = (const float*)d_in[28];
  const float* W_out = (const float*)d_in[29];

  const int Nn  = in_sizes[0] / HID;      // 20000 nodes
  const int Ee  = in_sizes[2] / 2;        // 320000 edges
  const int Fe  = in_sizes[1] / Ee;       // 64 edge features
  const int Elg = in_sizes[4] / 2;        // 600000 line-graph edges
  const int EG  = Elg + Ee;               // + self loops
  const int OUTD = 40;

  const int* H0  = Hm;       const int* H1  = Hm + Ee;
  const int* src = rei;      const int* dst = rei + Ee;
  const int* ls  = lg;       const int* ld  = lg + Elg;

  // ---- bump allocator over d_ws ----
  char* base = (char*)d_ws;
  size_t off = 0;
  auto take = [&](size_t elems) -> float* {
    float* p = (float*)(base + off);
    off += ((elems * sizeof(float) + 255) / 256) * 256;
    return p;
  };
  float* hbuf  = take((size_t)Ee * HID);   // h, later he
  float* gout  = take((size_t)Ee * HID);   // gat_out
  float* hs    = take(Ee);
  float* hd    = take(Ee);
  float* scg   = take(EG);                 // GAT scores -> ex -> alpha
  float* mbuf  = take(Ee);                 // segment max (reused for mix)
  float* sbuf  = take(Ee);                 // segment sum (reused for mix)
  float* accN  = take((size_t)Nn * HID);
  float* degN  = take(Nn);
  float* erep  = take((size_t)Nn * HID);
  float* degR  = take(Nn);
  float* meanb = take((size_t)Nn * HID);
  float* te1   = take((size_t)Nn * HID);
  float* aggr  = take((size_t)Nn * HID);
  float* nA    = take((size_t)Nn * HID);
  float* nB    = take((size_t)Nn * HID);
  float* nrep  = take((size_t)Nn * HID);
  float* qb    = take((size_t)Nn * HID);
  float* kb    = take((size_t)Nn * HID);
  float* vb    = take((size_t)Nn * HID);
  float* scE   = take(Ee);
  float* mixed = take((size_t)Nn * HID);
  float* logits= take((size_t)Nn * OUTD);
  (void)ws_size; (void)n_in; (void)out_size;

  const int EB = 4096;
  auto gemm = [&](const float* A, const float* B, float* C, const float* bias,
                  int M, int Ncol, int K, int addC) {
    dim3 g((Ncol + TN - 1) / TN, (M + TM - 1) / TM);
    if ((Ncol % TN) == 0)
      gemm_wmma_kernel<true><<<g, 256, 0, stream>>>(A, B, C, bias, M, Ncol, K, addC);
    else
      gemm_wmma_kernel<false><<<g, 256, 0, stream>>>(A, B, C, bias, M, Ncol, K, addC);
  };
  auto fill = [&](float* p, float v, size_t n) {
    fill_kernel<<<EB, 256, 0, stream>>>(p, v, n);
  };

  // ================= EdgeGCNV2: GAT over line graph =================
  gemm(et, W_gat, hbuf, nullptr, Ee, HID, Fe, 0);                       // h = et @ W_gat
  dual_dot_kernel<<<2048, 256, 0, stream>>>(hbuf, a_src, a_dst, hs, hd, Ee);
  gat_score_kernel<<<EB, 256, 0, stream>>>(hs, hd, ls, ld, scg, Elg, EG);
  fill(mbuf, -3.0e38f, Ee);
  fill(sbuf, 0.0f, Ee);
  seg_max_kernel<<<EB, 256, 0, stream>>>(scg, ld, mbuf, Elg, EG);
  seg_exp_kernel<<<EB, 256, 0, stream>>>(scg, ld, mbuf, sbuf, Elg, EG);
  seg_norm_kernel<<<EB, 256, 0, stream>>>(scg, ld, sbuf, Elg, EG);
  bcast_rows_kernel<<<EB, 256, 0, stream>>>(gout, b_gat, (size_t)Ee);
  weighted_scatter_kernel<<<8192, 256, 0, stream>>>(scg, hbuf, ls, ld, gout, Elg, EG);

  // ================= Edge -> Node incidence aggregation =================
  gemm(gout, W_etn, hbuf, nullptr, Ee, HID, HID, 0);                     // he (reuse hbuf)
  fill(accN, 0.0f, (size_t)Nn * HID);
  fill(degN, 0.0f, Nn);
  inc_scatter_kernel<<<8192, 256, 0, stream>>>(hbuf, H0, H1, accN, Ee);
  inc_deg_kernel<<<EB, 256, 0, stream>>>(H0, H1, degN, Ee);
  etn_lrelu_kernel<<<EB, 256, 0, stream>>>((const float4*)accN, degN, (float4*)accN, (size_t)Nn);
  gemm(accN, W_eg, erep, nullptr, Nn, HID, HID, 0);                      // edge_repr

  // degree over raw graph (shared by all SAGE layers)
  fill(degR, 0.0f, Nn);
  deg_count_kernel<<<EB, 256, 0, stream>>>(dst, degR, Ee);

  auto sage = [&](const float* in, const float* Wr, const float* Wn, const float* b,
                  float* out, int relu) {
    fill(meanb, 0.0f, (size_t)Nn * HID);
    sage_scatter_kernel<<<8192, 256, 0, stream>>>(in, src, dst, meanb, Ee);
    mean_div_kernel<<<EB, 256, 0, stream>>>((float4*)meanb, degR, (size_t)Nn);
    gemm(in, Wr, out, b, Nn, HID, HID, 0);
    gemm(meanb, Wn, out, nullptr, Nn, HID, HID, 1);
    if (relu) relu_kernel<<<EB, 256, 0, stream>>>((float4*)out, (size_t)Nn * HID / 4);
  };

  // edge_aggr SAGE (2 layers)
  sage(erep, Wr_e1, Wn_e1, b_e1, te1, 1);
  sage(te1,  Wr_e2, Wn_e2, b_e2, aggr, 0);
  // attr_node SAGE (3 layers)
  sage(x,  Wr_n1, Wn_n1, b_n1, nA, 1);
  sage(nA, Wr_n2, Wn_n2, b_n2, nB, 1);
  sage(nB, Wr_n3, Wn_n3, b_n3, nrep, 0);

  // ================= MixAttention =================
  gemm(nrep, Wq, qb, nullptr, Nn, HID, HID, 0);
  gemm(aggr, Wk, kb, nullptr, Nn, HID, HID, 0);
  gemm(aggr, Wv, vb, nullptr, Nn, HID, HID, 0);
  attn_score_kernel<<<8192, 256, 0, stream>>>(qb, kb, src, dst, scE, Ee);
  fill(mbuf, -3.0e38f, Nn);
  fill(sbuf, 0.0f, Nn);
  seg_max_kernel<<<EB, 256, 0, stream>>>(scE, dst, mbuf, Ee, Ee);
  seg_exp_kernel<<<EB, 256, 0, stream>>>(scE, dst, mbuf, sbuf, Ee, Ee);
  seg_norm_kernel<<<EB, 256, 0, stream>>>(scE, dst, sbuf, Ee, Ee);
  hipMemcpyAsync(mixed, nrep, (size_t)Nn * HID * sizeof(float),
                 hipMemcpyDeviceToDevice, stream);
  weighted_scatter_kernel<<<8192, 256, 0, stream>>>(scE, vb, src, dst, mixed, Ee, Ee);

  // ================= Output =================
  gemm(mixed, W_out, logits, nullptr, Nn, OUTD, HID, 0);
  logsoftmax40_kernel<<<2500, 256, 0, stream>>>(logits, (float*)d_out, Nn);
}
